// SceneFlowEstimatorPointConv_5291399708685
// MI455X (gfx1250) — matrix operand order repaired and checked by reading, not compiled, not run on hardware
//
#include <hip/hip_runtime.h>
#include <hip/hip_bf16.h>
#include <stdint.h>

#define BB        2
#define NP        8192
#define KNN       9
#define LEAKY_S   0.1f
#define BN_EPS    1e-5f

// ---- WMMA vector types (CDNA5 / gfx1250, wave32) ----
typedef __attribute__((ext_vector_type(16))) __bf16 v16bf;
typedef __attribute__((ext_vector_type(8)))  __bf16 v8bf;
typedef __attribute__((ext_vector_type(8)))  float  v8f;

__device__ __forceinline__ __bf16 f2bf(float f) {
  union { float f; unsigned u; } a; a.f = f;
  unsigned r = a.u + 0x7FFFu + ((a.u >> 16) & 1u);   // round-to-nearest-even
  union { unsigned short s; __bf16 b; } o; o.s = (unsigned short)(r >> 16);
  return o.b;
}

__device__ __forceinline__ float leakyf(float v) { return v > 0.f ? v : LEAKY_S * v; }

// =====================================================================
// 1) Brute-force KNN: one thread per query, LDS-tiled candidates,
//    fully unrolled register top-9 (ties keep lower index, ascending scan)
// =====================================================================
__global__ __launch_bounds__(256) void knn_kernel(const float* __restrict__ xyz,
                                                  int* __restrict__ idx) {
  const int b = blockIdx.y;
  const int n = blockIdx.x * 256 + threadIdx.x;
  const float* X = xyz + (size_t)b * 3 * NP;
  const float qx = X[n], qy = X[NP + n], qz = X[2 * NP + n];

  float bd[KNN]; int bi[KNN];
#pragma unroll
  for (int i = 0; i < KNN; ++i) { bd[i] = 3.4e38f; bi[i] = 0; }

  __shared__ float sx[256], sy[256], sz[256];
  for (int tile = 0; tile < NP / 256; ++tile) {
    const int c0 = tile * 256 + threadIdx.x;
    sx[threadIdx.x] = X[c0];
    sy[threadIdx.x] = X[NP + c0];
    sz[threadIdx.x] = X[2 * NP + c0];
    __syncthreads();
#pragma unroll 4
    for (int j = 0; j < 256; ++j) {
      const float dx = sx[j] - qx, dy = sy[j] - qy, dz = sz[j] - qz;
      const float d = dx * dx + dy * dy + dz * dz;
      if (d < bd[KNN - 1]) {
        bd[KNN - 1] = d; bi[KNN - 1] = tile * 256 + j;
#pragma unroll
        for (int s = KNN - 1; s > 0; --s) {
          if (bd[s] < bd[s - 1]) {
            float td = bd[s]; bd[s] = bd[s - 1]; bd[s - 1] = td;
            int   ti = bi[s]; bi[s] = bi[s - 1]; bi[s - 1] = ti;
          }
        }
      }
    }
    __syncthreads();
  }
#pragma unroll
  for (int k = 0; k < KNN; ++k) idx[((size_t)b * NP + n) * KNN + k] = bi[k];
}

// =====================================================================
// 2) Pack lin_w (fp32 [128 x Kin]) -> bf16 B-fragments for
//    v_wmma_f32_16x16x32_bf16: lane l (col = l&15, K-half = l>>4) holds
//    16 consecutive K values. Zero-pad K beyond Kin.
// =====================================================================
__global__ __launch_bounds__(256) void pack_w_kernel(const float* __restrict__ lin,
                                                     __bf16* __restrict__ Wp,
                                                     int nchunks, int Kin) {
  const int o = blockIdx.x * 256 + threadIdx.x;
  const int total = 8 * nchunks * 512;        // 8 col-tiles * nchunks * 32 lanes * 16
  if (o >= total) return;
  const int j  = o & 15;
  const int l  = (o >> 4) & 31;
  const int kc = (o >> 9) % nchunks;
  const int t  = o / (512 * nchunks);
  const int n  = t * 16 + (l & 15);
  const int k  = kc * 32 + (l >> 4) * 16 + j;
  const float v = (k < Kin) ? lin[(size_t)n * Kin + k] : 0.f;
  Wp[o] = f2bf(v);
}

// =====================================================================
// 3) Fold BatchNorm + linear bias: out = dot*scale + shift
// =====================================================================
__global__ void bnfold_kernel(const float* __restrict__ g, const float* __restrict__ bb,
                              const float* __restrict__ m, const float* __restrict__ v,
                              const float* __restrict__ lb,
                              float* __restrict__ scale, float* __restrict__ shift) {
  const int i = threadIdx.x;              // 128 channels
  const float s = g[i] * rsqrtf(v[i] + BN_EPS);
  scale[i] = s;
  shift[i] = (lb[i] - m[i]) * s + bb[i];
}

// =====================================================================
// 4) Fused PointConv: 16 points/block, 8 waves, bf16 WMMA GEMM in LDS
// =====================================================================
#define PC_PTS      16
#define CHUNKS_PASS 66
#define AGG_ROW     (CHUNKS_PASS * 32)                 // 2112 bf16 per point row
#define SM_AGG_B    (PC_PTS * AGG_ROW * 2)             // 67584
#define SM_FEAT_O   SM_AGG_B
#define SM_FEAT_B   (PC_PTS * KNN * 32 * 4)            // 18432
#define SM_WGT_O    (SM_FEAT_O + SM_FEAT_B)
#define SM_WGT_B    (PC_PTS * KNN * 16 * 4)            // 9216
#define SM_NRM_O    (SM_WGT_O + SM_WGT_B)
#define SM_NRM_B    (PC_PTS * KNN * 3 * 4)             // 1728
#define SM_NBR_O    (SM_NRM_O + SM_NRM_B)
#define SM_NBR_B    (PC_PTS * KNN * 4)                 // 576
#define SM_TOTAL    (SM_NBR_O + SM_NBR_B)              // 97536 bytes

__global__ __launch_bounds__(256) void pointconv_kernel(
    const float* __restrict__ xyz, const int* __restrict__ idx,
    const float* __restrict__ s0, int c0,
    const float* __restrict__ s1, int c1,
    const float* __restrict__ s2, int c2,
    const float* __restrict__ w0, const float* __restrict__ b0,
    const float* __restrict__ w1, const float* __restrict__ b1,
    const float* __restrict__ w2, const float* __restrict__ b2,
    const __bf16* __restrict__ Wp, int nchunks, int C,
    const float* __restrict__ scale, const float* __restrict__ shift,
    float* __restrict__ out) {
  extern __shared__ unsigned char smem[];
  __bf16* agg  = (__bf16*)smem;
  float*  feat = (float*)(smem + SM_FEAT_O);
  float*  wgt  = (float*)(smem + SM_WGT_O);
  float*  nrm  = (float*)(smem + SM_NRM_O);
  int*    nbr  = (int*)(smem + SM_NBR_O);

  const int b   = blockIdx.y;
  const int pt0 = blockIdx.x * PC_PTS;
  const int tid = threadIdx.x;

  // ---- Phase 0: neighbor indices, relative coords, WeightNet (3->8->8->16)
  if (tid < PC_PTS * KNN) {
    const int p = tid / KNN, k = tid % KNN;
    const int gp = pt0 + p;
    const int m  = idx[((size_t)b * NP + gp) * KNN + k];
    nbr[p * KNN + k] = m;
    const float* X = xyz + (size_t)b * 3 * NP;
    const float n0 = X[m] - X[gp];
    const float n1 = X[NP + m] - X[NP + gp];
    const float n2 = X[2 * NP + m] - X[2 * NP + gp];
    nrm[(p * KNN + k) * 3 + 0] = n0;
    nrm[(p * KNN + k) * 3 + 1] = n1;
    nrm[(p * KNN + k) * 3 + 2] = n2;
    float h0[8], h1[8];
#pragma unroll
    for (int o = 0; o < 8; ++o) {
      float a = fmaf(n0, w0[o * 3], fmaf(n1, w0[o * 3 + 1], fmaf(n2, w0[o * 3 + 2], b0[o])));
      h0[o] = fmaxf(a, 0.f);
    }
#pragma unroll
    for (int o = 0; o < 8; ++o) {
      float a = b1[o];
#pragma unroll
      for (int i = 0; i < 8; ++i) a = fmaf(h0[i], w1[o * 8 + i], a);
      h1[o] = fmaxf(a, 0.f);
    }
#pragma unroll
    for (int o = 0; o < 16; ++o) {
      float a = b2[o];
#pragma unroll
      for (int i = 0; i < 8; ++i) a = fmaf(h1[i], w2[o * 8 + i], a);
      wgt[(p * KNN + k) * 16 + o] = fmaxf(a, 0.f);
    }
  }
  __syncthreads();

  // per-thread (point, w-column) assignment for the agg einsum
  const int pidx = tid >> 4, wcol = tid & 15;
  float wreg[KNN];
#pragma unroll
  for (int k = 0; k < KNN; ++k) wreg[k] = wgt[(pidx * KNN + k) * 16 + wcol];

  // WMMA lane bookkeeping
  const int lane = tid & 31, wave = tid >> 5;
  const int hi   = lane >> 4;
  v8f acc{};   // persists across K passes

  const __bf16 zbf = f2bf(0.f);

  for (int pcBase = 0; pcBase < nchunks; pcBase += CHUNKS_PASS) {
    const int passChunks = min(CHUNKS_PASS, nchunks - pcBase);
    const int chBase = pcBase * 2;                  // 2 feature channels per 32-wide K chunk
    const int passCh = passChunks * 2;

    // zero agg tile (covers zero-padded K tail)
    for (int e = tid; e < PC_PTS * AGG_ROW; e += 256) agg[e] = zbf;
    __syncthreads();

    // ---- Phase 1: gather + einsum into bf16 agg tile, 32 channels/stage
    for (int sub = 0; sub < passCh; sub += 32) {
      const int base_c = chBase + sub;
      for (int e = tid; e < PC_PTS * KNN * 32; e += 256) {
        const int p = e / (KNN * 32);
        const int r = e - p * (KNN * 32);
        const int k = r >> 5, cc = r & 31;
        const int c = base_c + cc;
        float v = 0.f;
        if (c < C) {
          if (c < 3) {
            v = nrm[(p * KNN + k) * 3 + c];
          } else {
            int cp = c - 3;
            const int m = nbr[p * KNN + k];
            if (cp < c0) v = s0[((size_t)b * c0 + cp) * NP + m];
            else {
              cp -= c0;
              if (cp < c1) v = s1[((size_t)b * c1 + cp) * NP + m];
              else { cp -= c1; v = s2[((size_t)b * c2 + cp) * NP + m]; }
            }
          }
        }
        feat[e] = v;
      }
      __syncthreads();

      int lim = C - base_c; lim = lim < 0 ? 0 : (lim > 32 ? 32 : lim);
      for (int cc = 0; cc < lim; ++cc) {
        float a = 0.f;
#pragma unroll
        for (int k = 0; k < KNN; ++k)
          a = fmaf(feat[pidx * (KNN * 32) + k * 32 + cc], wreg[k], a);
        agg[pidx * AGG_ROW + (sub + cc) * 16 + wcol] = f2bf(a);
      }
      __syncthreads();
    }

    // ---- Phase 2: bf16 WMMA over this pass's K chunks
    // A (16x32): lane row = lane&15; elems 0-15 -> K base+hi*16+s (matches B packing).
    // B pre-packed for contiguous 32B loads.
    const int arow = lane & 15;
    for (int kc = 0; kc < passChunks; ++kc) {
      const __bf16* ap = agg + arow * AGG_ROW + kc * 32;
      union { v16bf v; v8bf h[2]; } A;
      A.h[0] = *(const v8bf*)(ap + hi * 16);
      A.h[1] = *(const v8bf*)(ap + hi * 16 + 8);
      const __bf16* bp = Wp + (((size_t)wave * nchunks + (pcBase + kc)) * 32 + lane) * 16;
      const v16bf Bv = *(const v16bf*)bp;
      acc = __builtin_amdgcn_wmma_f32_16x16x32_bf16(false, A.v, false, Bv,
                                                    (short)0, acc, false, false);
    }
    __syncthreads();
  }

  // ---- Epilogue: D layout -> col = lane&15, rows = hi*8 + r (contiguous pts)
  const int col = lane & 15;
  const int co  = wave * 16 + col;
  const float sC = scale[co], sH = shift[co];
  float vals[8];
#pragma unroll
  for (int r = 0; r < 8; ++r) vals[r] = leakyf(acc[r] * sC + sH);
  float* op = out + ((size_t)b * 128 + co) * NP + pt0 + hi * 8;
  *(float4*)(op)     = make_float4(vals[0], vals[1], vals[2], vals[3]);
  *(float4*)(op + 4) = make_float4(vals[4], vals[5], vals[6], vals[7]);
}

// =====================================================================
// 5) MLP head: per-(b,n,o) dot product, channel-major coalesced loads.
//    mode 0 = none, 1 = leaky, 2 = clip(-200,200)
// =====================================================================
__global__ __launch_bounds__(256) void mlp_kernel(const float* __restrict__ in,
                                                  const float* __restrict__ W,
                                                  const float* __restrict__ bias,
                                                  float* __restrict__ out,
                                                  int Cin, int Cout, int mode) {
  const int n = blockIdx.x * 256 + threadIdx.x;
  const int o = blockIdx.y, b = blockIdx.z;
  const float* x = in + (size_t)b * Cin * NP + n;
  const float* w = W + (size_t)o * Cin;
  float acc = bias[o];
  for (int c = 0; c < Cin; ++c) acc = fmaf(x[(size_t)c * NP], w[c], acc);
  if (mode == 1)      acc = leakyf(acc);
  else if (mode == 2) acc = fminf(fmaxf(acc, -200.f), 200.f);
  out[((size_t)b * Cout + o) * NP + n] = acc;
}

// =====================================================================
// Host launch
// =====================================================================
extern "C" void kernel_launch(void* const* d_in, const int* in_sizes, int n_in,
                              void* d_out, int out_size, void* d_ws, size_t ws_size,
                              hipStream_t stream) {
  (void)in_sizes; (void)n_in; (void)out_size; (void)ws_size;
  auto F = [&](int i) { return (const float*)d_in[i]; };

  const float* xyz   = F(0);
  const float* feats = F(1);
  const float* cost  = F(2);
  const float* flow  = F(3);
  // pc layers: base 4 and 16
  const int L0 = 4, L1 = 16;
  const float* mlp0_w = F(28); const float* mlp0_b = F(29);
  const float* mlp1_w = F(30); const float* mlp1_b = F(31);
  const float* fc_w   = F(32); const float* fc_b   = F(33);

  // workspace partition (256B aligned)
  uint8_t* ws = (uint8_t*)d_ws; size_t off = 0;
  auto take = [&](size_t bytes) -> void* {
    void* p = ws + off; off += (bytes + 255) & ~(size_t)255; return p;
  };
  int*    idxws = (int*)   take((size_t)BB * NP * KNN * 4);
  float*  pts1  = (float*) take((size_t)BB * 128 * NP * 4);
  float*  pts2  = (float*) take((size_t)BB * 128 * NP * 4);
  float*  h0    = (float*) take((size_t)BB * 128 * NP * 4);
  __bf16* Wp0   = (__bf16*)take((size_t)131 * 8 * 512 * 2);   // 4192x128 bf16
  __bf16* Wp1   = (__bf16*)take((size_t)66  * 8 * 512 * 2);   // 2112x128 bf16 (padded)
  float*  sc0   = (float*) take(128 * 4);
  float*  sh0   = (float*) take(128 * 4);
  float*  sc1   = (float*) take(128 * 4);
  float*  sh1   = (float*) take(128 * 4);

  // weight packing + BN folding (tiny)
  {
    int tot0 = 8 * 131 * 512, tot1 = 8 * 66 * 512;
    pack_w_kernel<<<(tot0 + 255) / 256, 256, 0, stream>>>(F(L0 + 6), Wp0, 131, 4192);
    pack_w_kernel<<<(tot1 + 255) / 256, 256, 0, stream>>>(F(L1 + 6), Wp1, 66, 2096);
    bnfold_kernel<<<1, 128, 0, stream>>>(F(L0 + 8), F(L0 + 9), F(L0 + 10), F(L0 + 11),
                                         F(L0 + 7), sc0, sh0);
    bnfold_kernel<<<1, 128, 0, stream>>>(F(L1 + 8), F(L1 + 9), F(L1 + 10), F(L1 + 11),
                                         F(L1 + 7), sc1, sh1);
  }

  // KNN
  knn_kernel<<<dim3(NP / 256, BB), 256, 0, stream>>>(xyz, idxws);

  // PointConv layer 0: sources feats(128) | cost(128) | flow(3), C = 262
  pointconv_kernel<<<dim3(NP / PC_PTS, BB), 256, SM_TOTAL, stream>>>(
      xyz, idxws, feats, 128, cost, 128, flow, 3,
      F(L0 + 0), F(L0 + 1), F(L0 + 2), F(L0 + 3), F(L0 + 4), F(L0 + 5),
      Wp0, 131, 262, sc0, sh0, pts1);

  // PointConv layer 1: source pts1(128), C = 131
  pointconv_kernel<<<dim3(NP / PC_PTS, BB), 256, SM_TOTAL, stream>>>(
      xyz, idxws, pts1, 128, (const float*)nullptr, 0, (const float*)nullptr, 0,
      F(L1 + 0), F(L1 + 1), F(L1 + 2), F(L1 + 3), F(L1 + 4), F(L1 + 5),
      Wp1, 66, 131, sc1, sh1, pts2);

  // MLP head
  float* outPts  = (float*)d_out;                       // [B,64,N]
  float* outFlow = outPts + (size_t)BB * 64 * NP;       // [B,3,N]
  mlp_kernel<<<dim3(NP / 256, 128, BB), 256, 0, stream>>>(pts2, mlp0_w, mlp0_b, h0, 128, 128, 1);
  mlp_kernel<<<dim3(NP / 256, 64,  BB), 256, 0, stream>>>(h0, mlp1_w, mlp1_b, outPts, 128, 64, 1);
  mlp_kernel<<<dim3(NP / 256, 3,   BB), 256, 0, stream>>>(outPts, fc_w, fc_b, outFlow, 64, 3, 2);
}
